// Attention_25340307046887
// MI455X (gfx1250) — matrix operand order repaired and checked by reading, not compiled, not run on hardware
//
#include <hip/hip_runtime.h>

typedef __attribute__((ext_vector_type(16))) __bf16 bf16x16;
typedef __attribute__((ext_vector_type(8)))  __bf16 bf16x8;
typedef __attribute__((ext_vector_type(4)))  __bf16 bf16x4;
typedef __attribute__((ext_vector_type(8)))  float  floatx8;
typedef __attribute__((ext_vector_type(4)))  float  floatx4;

#define WMMA_BF16(A, B, C) \
  __builtin_amdgcn_wmma_f32_16x16x32_bf16(false, (A), false, (B), (short)0, (C), false, false)

// ---------------------------------------------------------------------------
// fp32 -> bf16 bulk convert (used once for the 4 weight matrices; L2-resident
// afterwards, so every GEMM B-fragment becomes two plain global_load_b128).
// ---------------------------------------------------------------------------
__global__ __launch_bounds__(256) void cvt_bf16_kernel(
    const float* __restrict__ src, __bf16* __restrict__ dst, int n4) {
  const int i = (blockIdx.x * 256 + threadIdx.x);
  if (i < n4) {
    const floatx4 f = *(const floatx4*)(src + (size_t)i * 4);
    bf16x4 b;
    b[0] = (__bf16)f[0]; b[1] = (__bf16)f[1];
    b[2] = (__bf16)f[2]; b[3] = (__bf16)f[3];
    *(bf16x4*)(dst + (size_t)i * 4) = b;
  }
}

// ---------------------------------------------------------------------------
// C[m,e] = sum_k A[m,k] * Wb[e,k] + bias[e]    (torch Linear: x @ W.T + b)
// A: M x K (row-major, TIn = float or __bf16), Wb: E x K bf16, C: M x E fp32.
// Block: 256 threads = 8 waves. Wave tile 16(M) x 64(E). Block tile 64 x 128.
// ---------------------------------------------------------------------------
template <typename TIn>
__global__ __launch_bounds__(256, 1) void linear_wmma(
    const TIn* __restrict__ A,
    const __bf16* __restrict__ Wb,
    const float* __restrict__ bias,
    float* __restrict__ Cout,
    int M, int E, int K) {
  const int wave = threadIdx.x >> 5;
  const int lane = threadIdx.x & 31;
  const int half = lane >> 4;
  const int l16  = lane & 15;

  const int mBase = blockIdx.x * 64  + (wave & 3) * 16;
  const int eBase = blockIdx.y * 128 + (wave >> 2) * 64;

  const TIn* arow = A + (size_t)(mBase + l16) * K;

  floatx8 acc[4];
  for (int s = 0; s < 4; ++s)
    for (int i = 0; i < 8; ++i) acc[s][i] = 0.0f;

  for (int k0 = 0; k0 < K; k0 += 32) {
    // L2 prefetch of upcoming A data (gfx1250 global_prefetch_b8)
    __builtin_prefetch(arow + k0 + 64, 0, 0);

    // A fragment: contiguous k order (k = 16*half + s, as in B) — lane holds row (l16), K chunks at
    // 16*half and 16*half + 8 (8 contiguous values each).
    bf16x16 af;
    if constexpr (sizeof(TIn) == 2) {
      const bf16x8 a0 = *(const bf16x8*)((const __bf16*)arow + k0 + 16 * half);
      const bf16x8 a1 = *(const bf16x8*)((const __bf16*)arow + k0 + 16 * half + 8);
      for (int i = 0; i < 8; ++i) { af[i] = a0[i]; af[8 + i] = a1[i]; }
    } else {
      const TIn* p0 = arow + k0 + 16 * half;
      const TIn* p1 = arow + k0 + 16 * half + 8;
      for (int i = 0; i < 8; ++i) {
        af[i]     = (__bf16)(float)p0[i];
        af[8 + i] = (__bf16)(float)p1[i];
      }
    }

    for (int s = 0; s < 4; ++s) {
      // B fragment: 32x16 layout — lane holds column (eBase+s*16+l16),
      // 16 contiguous K values at k0 + 16*half. Column e of B is row e of W
      // (K-contiguous) since B[k][e] = W[e][k]. Direct 32B bf16 load.
      const __bf16* wrow = Wb + (size_t)(eBase + s * 16 + l16) * K + k0 + 16 * half;
      const bf16x16 bf = *(const bf16x16*)wrow;
      acc[s] = WMMA_BF16(af, bf, acc[s]);
    }
  }

  // C/D layout: VGPR i holds row (i + 8*half), column l16.
  for (int s = 0; s < 4; ++s) {
    const int col = eBase + s * 16 + l16;
    const float bv = bias[col];
    for (int i = 0; i < 8; ++i) {
      const int row = mBase + i + 8 * half;
      Cout[(size_t)row * E + col] = acc[s][i] + bv;
    }
  }
}

// ---------------------------------------------------------------------------
// Flash attention per (n, h, 128 q-rows). qh/kh/vh fp32 [N, L, H, 64].
// Each of 8 waves owns a 16-row q tile; streams kv in chunks of 32 staged in
// LDS (K row-major, V transposed). Writes bf16 mix [N, L, H*64].
// ---------------------------------------------------------------------------
__global__ __launch_bounds__(256, 1) void attn_wmma(
    const float* __restrict__ qh,
    const float* __restrict__ kh,
    const float* __restrict__ vh,
    __bf16* __restrict__ mixb) {
  const int L = 1024, H = 12, Cd = 64;
  const int n = blockIdx.z;
  const int h = blockIdx.y;
  const int qBase = blockIdx.x * 128;

  const int wave = threadIdx.x >> 5;
  const int lane = threadIdx.x & 31;
  const int half = lane >> 4;
  const int l16  = lane & 15;

  __shared__ __align__(32) __bf16 ldsK[32 * 64];    // [kv][c]
  __shared__ __align__(32) __bf16 ldsVt[64 * 32];   // [c][kv] (transposed)
  __shared__ __align__(32) __bf16 ldsP[8][16 * 32]; // per-wave P tile [row][kv]

  // Q fragments (C = 64 -> two K=32 fragments), softmax scale folded in.
  const float scale = 0.125f; // 1/sqrt(64)
  const int qrow = qBase + wave * 16 + l16;
  const float* qp = qh + (((size_t)n * L + qrow) * H + h) * Cd;
  bf16x16 q0, q1;
  for (int i = 0; i < 8; ++i) {
    q0[i]     = (__bf16)(qp[0  + 16 * half + i] * scale);
    q0[8 + i] = (__bf16)(qp[8  + 16 * half + i] * scale);
    q1[i]     = (__bf16)(qp[32 + 16 * half + i] * scale);
    q1[8 + i] = (__bf16)(qp[40 + 16 * half + i] * scale);
  }

  float mrun[8], lrun[8];
  floatx8 o[4];
  for (int i = 0; i < 8; ++i) { mrun[i] = -1.0e30f; lrun[i] = 0.0f; }
  for (int s = 0; s < 4; ++s)
    for (int i = 0; i < 8; ++i) o[s][i] = 0.0f;

  for (int kv0 = 0; kv0 < L; kv0 += 32) {
    // Cooperative stage of K (row-major) and V (transposed) tiles, fp32->bf16.
    {
      const int kv = threadIdx.x >> 3;        // 0..31
      const int c0 = (threadIdx.x & 7) * 8;   // 0..56
      const float* kp = kh + (((size_t)n * L + kv0 + kv) * H + h) * Cd + c0;
      const float* vp = vh + (((size_t)n * L + kv0 + kv) * H + h) * Cd + c0;
      for (int i = 0; i < 8; ++i) {
        ldsK[kv * 64 + c0 + i]    = (__bf16)kp[i];
        ldsVt[(c0 + i) * 32 + kv] = (__bf16)vp[i];
      }
    }
    __syncthreads();

    // S = (Q*scale) @ K^T : two 16x16 kv tiles, K-dim = 64 (2 WMMAs each).
    floatx8 s0, s1;
    for (int i = 0; i < 8; ++i) { s0[i] = 0.0f; s1[i] = 0.0f; }
    {
      const bf16x16 b00 = *(const bf16x16*)&ldsK[(0  + l16) * 64 + 0  + 16 * half];
      const bf16x16 b10 = *(const bf16x16*)&ldsK[(16 + l16) * 64 + 0  + 16 * half];
      s0 = WMMA_BF16(q0, b00, s0);
      s1 = WMMA_BF16(q0, b10, s1);
      const bf16x16 b01 = *(const bf16x16*)&ldsK[(0  + l16) * 64 + 32 + 16 * half];
      const bf16x16 b11 = *(const bf16x16*)&ldsK[(16 + l16) * 64 + 32 + 16 * half];
      s0 = WMMA_BF16(q1, b01, s0);
      s1 = WMMA_BF16(q1, b11, s1);
    }

    // Online softmax over this kv chunk. Row i+8*half lives on a 16-lane half;
    // reduce across columns with xor-shuffles (masks 1..8 stay in-half).
    for (int i = 0; i < 8; ++i) {
      float v = fmaxf(s0[i], s1[i]);
      v = fmaxf(v, __shfl_xor(v, 1));
      v = fmaxf(v, __shfl_xor(v, 2));
      v = fmaxf(v, __shfl_xor(v, 4));
      v = fmaxf(v, __shfl_xor(v, 8));
      const float mn = fmaxf(mrun[i], v);
      const float alpha = __expf(mrun[i] - mn);
      mrun[i] = mn;
      lrun[i] *= alpha;
      o[0][i] *= alpha; o[1][i] *= alpha; o[2][i] *= alpha; o[3][i] *= alpha;

      const float p0 = __expf(s0[i] - mn);
      const float p1 = __expf(s1[i] - mn);
      const int row = i + 8 * half;
      ldsP[wave][row * 32 + l16]      = (__bf16)p0;
      ldsP[wave][row * 32 + 16 + l16] = (__bf16)p1;

      float ps = p0 + p1;
      ps += __shfl_xor(ps, 1);
      ps += __shfl_xor(ps, 2);
      ps += __shfl_xor(ps, 4);
      ps += __shfl_xor(ps, 8);
      lrun[i] += ps;
    }

    // Re-layout P (C-layout -> A-layout) through per-wave LDS scratch.
    bf16x16 pf;
    {
      const bf16x8 p0 = *(const bf16x8*)&ldsP[wave][l16 * 32 + 16 * half];
      const bf16x8 p1 = *(const bf16x8*)&ldsP[wave][l16 * 32 + 16 * half + 8];
      for (int i = 0; i < 8; ++i) { pf[i] = p0[i]; pf[8 + i] = p1[i]; }
    }

    // O += P @ V : 4 column tiles of 16 (head dim 64).
    for (int s = 0; s < 4; ++s) {
      const bf16x16 vb = *(const bf16x16*)&ldsVt[(s * 16 + l16) * 32 + 16 * half];
      o[s] = WMMA_BF16(pf, vb, o[s]);
    }
    __syncthreads();
  }

  // Normalize and write bf16 mix [N, L, H*64].
  for (int s = 0; s < 4; ++s) {
    for (int i = 0; i < 8; ++i) {
      const int row = qBase + wave * 16 + i + 8 * half;
      const int c   = s * 16 + l16;
      mixb[((size_t)n * L + row) * (H * Cd) + h * Cd + c] =
          (__bf16)(o[s][i] / lrun[i]);
    }
  }
}

// ---------------------------------------------------------------------------
// Launch. Inputs: q,k,v,Wq,bq,Wk,bk,Wv,bv,Wo,bo (all fp32).
// d_out = [ out | qh | kh | vh ], each N*L*D fp32.
// d_ws  = [ mix bf16 (N*L*D) | Wq_bf | Wk_bf | Wv_bf | Wo_bf ]  (~30 MB)
// ---------------------------------------------------------------------------
extern "C" void kernel_launch(void* const* d_in, const int* in_sizes, int n_in,
                              void* d_out, int out_size, void* d_ws, size_t ws_size,
                              hipStream_t stream) {
  (void)in_sizes; (void)n_in; (void)out_size; (void)ws_size;
  const int N = 16, L = 1024, D = 768;
  const size_t S1 = (size_t)N * L * D;     // 12,582,912
  const size_t WSZ = (size_t)D * D;        // 589,824

  const float* q  = (const float*)d_in[0];
  const float* k  = (const float*)d_in[1];
  const float* v  = (const float*)d_in[2];
  const float* Wq = (const float*)d_in[3];
  const float* bq = (const float*)d_in[4];
  const float* Wk = (const float*)d_in[5];
  const float* bk = (const float*)d_in[6];
  const float* Wv = (const float*)d_in[7];
  const float* bv = (const float*)d_in[8];
  const float* Wo = (const float*)d_in[9];
  const float* bo = (const float*)d_in[10];

  float* out = (float*)d_out;
  float* qh  = out + S1;
  float* kh  = out + 2 * S1;
  float* vh  = out + 3 * S1;

  __bf16* mixb = (__bf16*)d_ws;            // S1 bf16
  __bf16* Wqb  = mixb + S1;
  __bf16* Wkb  = Wqb + WSZ;
  __bf16* Wvb  = Wkb + WSZ;
  __bf16* Wob  = Wvb + WSZ;

  dim3 blk(256);

  // One-time weight conversion (bf16 weights then stay L2-resident).
  const int n4 = (int)(WSZ / 4);           // 147,456 -> grid 576
  dim3 gC((n4 + 255) / 256);
  cvt_bf16_kernel<<<gC, blk, 0, stream>>>(Wq, Wqb, n4);
  cvt_bf16_kernel<<<gC, blk, 0, stream>>>(Wk, Wkb, n4);
  cvt_bf16_kernel<<<gC, blk, 0, stream>>>(Wv, Wvb, n4);
  cvt_bf16_kernel<<<gC, blk, 0, stream>>>(Wo, Wob, n4);

  const int M = N * L;                     // 16384 rows
  dim3 gL(M / 64, D / 128);                // (256, 6)

  linear_wmma<float><<<gL, blk, 0, stream>>>(q, Wqb, bq, qh, M, D, D);
  linear_wmma<float><<<gL, blk, 0, stream>>>(k, Wkb, bk, kh, M, D, D);
  linear_wmma<float><<<gL, blk, 0, stream>>>(v, Wvb, bv, vh, M, D, D);

  dim3 gA(L / 128, 12, N);                 // (8, 12, 16)
  attn_wmma<<<gA, blk, 0, stream>>>(qh, kh, vh, mixb);

  linear_wmma<__bf16><<<gL, blk, 0, stream>>>(mixb, Wob, bo, out, M, D, D);
}
